// CausalSelfAttention_55465207660698
// MI455X (gfx1250) — matrix operand order repaired and checked by reading, not compiled, not run on hardware
//
#include <hip/hip_runtime.h>

typedef __attribute__((ext_vector_type(16))) __bf16 v16bf;
typedef __attribute__((ext_vector_type(8)))  float  v8f;
typedef __attribute__((ext_vector_type(4)))  unsigned int u32x4;

#define QSCALE 0.08838834764831845f  // 1/sqrt(128)

__device__ __forceinline__ unsigned int packbf2(float a, float b) {
  union { __bf16 h[2]; unsigned int u; } c;
  c.h[0] = (__bf16)a; c.h[1] = (__bf16)b;  // hardware f32->bf16 RNE
  return c.u;
}

__device__ __forceinline__ v8f wmma_bf16(v16bf a, v16bf b, v8f c) {
  return __builtin_amdgcn_wmma_f32_16x16x32_bf16(false, a, false, b, (short)0, c,
                                                 false, false);
}

// Assemble a 16-element bf16 fragment from two 16-byte LDS chunks (pure b128 loads).
__device__ __forceinline__ v16bf ld_frag(const unsigned short* p0,
                                         const unsigned short* p1) {
  union { u32x4 q[2]; v16bf v; } u;
  u.q[0] = *(const u32x4*)p0;
  u.q[1] = *(const u32x4*)p1;
  return u.v;
}

// A fragment 16x32 (MxK) from row-major [M][K] bf16 LDS tile.
// Per ISA: lane<16 -> M=lane, K chunks {hi*8..+7, 16+hi*8..+7}; both 16B-aligned.
__device__ __forceinline__ v16bf frag_a(const unsigned short* lds, int stride) {
  const int lane = threadIdx.x & 31;
  const unsigned short* r = lds + (lane & 15) * stride + ((lane >> 4) << 3);
  return ld_frag(r, r + 16);
}

// B fragment 32x16 (KxN) from K-contiguous [N][K] LDS tile: K chunks {hi*8..+7, 16+hi*8..+7}.
__device__ __forceinline__ v16bf frag_bt(const unsigned short* lds, int stride) {
  const int lane = threadIdx.x & 31;
  const unsigned short* r = lds + (lane & 15) * stride + ((lane >> 4) << 3);
  return ld_frag(r, r + 16);
}

// -------- KV-cache concat: streaming float4 copy of past rows 0..4095 --------
__global__ __launch_bounds__(256)
void copy_cache_kernel(const float4* __restrict__ src, float4* __restrict__ dst) {
  const int i = blockIdx.x * blockDim.x + threadIdx.x;  // < 16777216
  const int bh = i >> 17;                                // / (4096*128/4)
  const int pos = i & 131071;
  dst[(size_t)bh * 133120 + pos] = src[i];               // 4160*128/4 stride
}

// -------- bf16 WMMA GEMM, 64x128 tile, BK=32, 8 waves (4x2), 4 WMMA/wave/step ----
// mode 0: A=x, B=W_attn (N=6144); scatter q (scaled) / k_new / v_new
// mode 1: A=y_ws, B=W_proj (N=2048); plain store to yo
__global__ __launch_bounds__(256)
void gemm_bf16_kernel(const float* __restrict__ A, const float* __restrict__ B,
                      int K, int N, int mode,
                      float* __restrict__ qws, float* __restrict__ ko,
                      float* __restrict__ vo, float* __restrict__ yo) {
  const int AS = 40, BS = 40;  // strides (ushort), multiples of 8 -> 16B rows
  __shared__ __align__(16) unsigned short As[64 * 40];    // [M][K] bf16
  __shared__ __align__(16) unsigned short BsT[128 * 40];  // [N][K] bf16 (transposed)

  const int tid = threadIdx.x;
  const int lane = tid & 31, w = tid >> 5;
  const int wm = w >> 1, wn = w & 1;
  const int m0 = blockIdx.y * 64, n0 = blockIdx.x * 128;

  v8f acc[4] = {{}, {}, {}, {}};

  const int ar = tid >> 2, ac = (tid & 3) * 8;   // A stage: 64x32, 8 floats/thread
  const int kp = tid >> 4, nc = (tid & 15) * 8;  // B stage: 2 k-rows x 8 n/thread

  for (int k0 = 0; k0 < K; k0 += 32) {
    {  // A tile: pack 8 bf16 -> one b128 store
      const float* ap = A + (size_t)(m0 + ar) * K + k0 + ac;
      u32x4 pk;
      pk.x = packbf2(ap[0], ap[1]); pk.y = packbf2(ap[2], ap[3]);
      pk.z = packbf2(ap[4], ap[5]); pk.w = packbf2(ap[6], ap[7]);
      *(u32x4*)&As[ar * AS + ac] = pk;
    }
    {  // B tile transposed: pair K-adjacent rows -> 8 b32 stores
      const float* b0 = B + (size_t)(k0 + 2 * kp) * N + n0 + nc;
      const float* b1 = b0 + N;
      if (k0 + 32 < K) __builtin_prefetch(b0 + (size_t)32 * N, 0, 1);
#pragma unroll
      for (int i = 0; i < 8; ++i)
        *(unsigned int*)&BsT[(nc + i) * BS + 2 * kp] = packbf2(b0[i], b1[i]);
    }
    __syncthreads();

    v16bf a = frag_a(As + (wm * 16) * AS, AS);
#pragma unroll
    for (int nt = 0; nt < 4; ++nt) {
      v16bf b = frag_bt(BsT + (wn * 64 + nt * 16) * BS, BS);
      acc[nt] = wmma_bf16(a, b, acc[nt]);
    }
    __syncthreads();
  }

  const int n = lane & 15, hi = lane >> 4;
#pragma unroll
  for (int nt = 0; nt < 4; ++nt) {
#pragma unroll
    for (int r = 0; r < 8; ++r) {
      const float val = acc[nt][r];
      const int row_g = m0 + wm * 16 + hi * 8 + r;
      const int col_g = n0 + wn * 64 + nt * 16 + n;
      if (mode == 0) {
        const int b = row_g >> 6, t = row_g & 63;
        const int sec = col_g >> 11, cc = col_g & 2047;
        const int h = cc >> 7, d = cc & 127;
        if (sec == 0)
          qws[(((size_t)b * 16 + h) * 64 + t) * 128 + d] = val * QSCALE;
        else if (sec == 1)
          ko[((size_t)b * 16 + h) * 4160 * 128 + (size_t)(4096 + t) * 128 + d] = val;
        else
          vo[((size_t)b * 16 + h) * 4160 * 128 + (size_t)(4096 + t) * 128 + d] = val;
      } else {
        yo[(size_t)row_g * N + col_g] = val;
      }
    }
  }
}

// -------- Flash attention: one block per (b,h), 8 waves, online softmax --------
__global__ __launch_bounds__(256)
void attn_kernel(const float* __restrict__ qws, const float* __restrict__ kc,
                 const float* __restrict__ vc, float* __restrict__ yws) {
  const int QS = 136, VS = 72, SS = 68, PS = 64;  // ushort/float strides
  __shared__ __align__(16) unsigned short Qs[64 * 136];   // Q [row][hd] bf16
  __shared__ __align__(16) unsigned short KVs[128 * 72];  // K [key][hd] (64*136) then V^T [hd][key]
  __shared__ __align__(16) float Ss[64 * 68];             // raw scores f32
  __shared__ __align__(16) unsigned short Ps[64 * 64];    // probs bf16
  __shared__ float mrow[64], lrow[64], arow[64];

  const int tid = threadIdx.x;
  const int lane = tid & 31, w = tid >> 5;
  const int wm = w >> 1, wn = w & 1;
  const int n = lane & 15, hi = lane >> 4;

  const int bh = blockIdx.x;
  const int b = bh >> 4, h = bh & 15;

  const float* qp = qws + (size_t)bh * 64 * 128;
  const float* kp = kc + (size_t)bh * 4160 * 128;
  const float* vp = vc + (size_t)bh * 4160 * 128;

  {  // stage Q (pre-scaled): 32 floats/thread -> 4 b128 stores
    const int r = tid >> 2, c0 = (tid & 3) * 32;
#pragma unroll
    for (int jj = 0; jj < 4; ++jj) {
      const float* s = qp + r * 128 + c0 + jj * 8;
      u32x4 pk;
      pk.x = packbf2(s[0], s[1]); pk.y = packbf2(s[2], s[3]);
      pk.z = packbf2(s[4], s[5]); pk.w = packbf2(s[6], s[7]);
      *(u32x4*)&Qs[r * QS + c0 + jj * 8] = pk;
    }
  }
  if (tid < 64) { mrow[tid] = -3.0e38f; lrow[tid] = 0.0f; }

  v8f o0 = {}, o1 = {}, o2 = {}, o3 = {};
  __syncthreads();

  for (int kb = 0; kb < 65; ++kb) {
    {  // stage K block [64 keys][128 hd], row-major (keys are B-rows => free transpose)
      const int r = tid >> 2, c0 = (tid & 3) * 32;
      const float* src = kp + (size_t)(kb * 64 + r) * 128 + c0;
#pragma unroll
      for (int jj = 0; jj < 4; ++jj) {
        const float* s = src + jj * 8;
        u32x4 pk;
        pk.x = packbf2(s[0], s[1]); pk.y = packbf2(s[2], s[3]);
        pk.z = packbf2(s[4], s[5]); pk.w = packbf2(s[6], s[7]);
        *(u32x4*)&KVs[r * QS + c0 + jj * 8] = pk;
      }
    }
    __syncthreads();

    // S = Q * K^T : each wave 16x32, contraction over hd=128
    v8f s0 = {}, s1 = {};
#pragma unroll
    for (int ks = 0; ks < 4; ++ks) {
      v16bf a  = frag_a(Qs + (wm * 16) * QS + ks * 32, QS);
      v16bf t0 = frag_bt(KVs + (wn * 32) * QS + ks * 32, QS);
      v16bf t1 = frag_bt(KVs + (wn * 32 + 16) * QS + ks * 32, QS);
      s0 = wmma_bf16(a, t0, s0);
      s1 = wmma_bf16(a, t1, s1);
    }
#pragma unroll
    for (int r = 0; r < 8; ++r) {
      const int row = wm * 16 + hi * 8 + r;
      Ss[row * SS + wn * 32 + n]      = s0[r];
      Ss[row * SS + wn * 32 + 16 + n] = s1[r];
    }
    __syncthreads();

    {  // stage V transposed [hd][key]: pair keys -> 16 b32 stores/thread
      const int hd0 = (tid & 7) * 16, kp2 = tid >> 3;
      const float* s0v = vp + (size_t)(kb * 64 + 2 * kp2) * 128 + hd0;
      const float* s1v = s0v + 128;
#pragma unroll
      for (int i = 0; i < 16; ++i)
        *(unsigned int*)&KVs[(hd0 + i) * VS + 2 * kp2] = packbf2(s0v[i], s1v[i]);
    }
    if (tid < 64) {  // online softmax, one row per thread, float4 scans
      const float4* row4 = (const float4*)&Ss[tid * SS];
      float bm = -3.0e38f;
#pragma unroll
      for (int j = 0; j < 16; ++j) {
        const float4 v4 = row4[j];
        bm = fmaxf(bm, fmaxf(fmaxf(v4.x, v4.y), fmaxf(v4.z, v4.w)));
      }
      const float mo = mrow[tid];
      const float mn = fmaxf(mo, bm);
      const float al = __expf(mo - mn);
      float sum = 0.0f;
#pragma unroll
      for (int j0 = 0; j0 < 8; ++j0) {
        const float4 a4 = row4[2 * j0];
        const float4 b4 = row4[2 * j0 + 1];
        const float p0 = __expf(a4.x - mn), p1 = __expf(a4.y - mn);
        const float p2 = __expf(a4.z - mn), p3 = __expf(a4.w - mn);
        const float p4 = __expf(b4.x - mn), p5 = __expf(b4.y - mn);
        const float p6 = __expf(b4.z - mn), p7 = __expf(b4.w - mn);
        sum += ((p0 + p1) + (p2 + p3)) + ((p4 + p5) + (p6 + p7));
        u32x4 pk;
        pk.x = packbf2(p0, p1); pk.y = packbf2(p2, p3);
        pk.z = packbf2(p4, p5); pk.w = packbf2(p6, p7);
        *(u32x4*)&Ps[tid * PS + j0 * 8] = pk;
      }
      mrow[tid] = mn;
      lrow[tid] = al * lrow[tid] + sum;
      arow[tid] = al;
    }
    __syncthreads();

    // O = alpha*O + P @ V : each wave 16 rows x 64 hd-cols
#pragma unroll
    for (int r = 0; r < 8; ++r) {
      const float al = arow[wm * 16 + hi * 8 + r];
      o0[r] *= al; o1[r] *= al; o2[r] *= al; o3[r] *= al;
    }
#pragma unroll
    for (int ks = 0; ks < 2; ++ks) {
      v16bf a  = frag_a(Ps + (wm * 16) * PS + ks * 32, PS);
      v16bf v0 = frag_bt(KVs + (wn * 64) * VS + ks * 32, VS);
      v16bf v1 = frag_bt(KVs + (wn * 64 + 16) * VS + ks * 32, VS);
      v16bf v2 = frag_bt(KVs + (wn * 64 + 32) * VS + ks * 32, VS);
      v16bf v3 = frag_bt(KVs + (wn * 64 + 48) * VS + ks * 32, VS);
      o0 = wmma_bf16(a, v0, o0);
      o1 = wmma_bf16(a, v1, o1);
      o2 = wmma_bf16(a, v2, o2);
      o3 = wmma_bf16(a, v3, o3);
    }
    __syncthreads();
  }

  // normalize and write y in [B, T, C] layout (C = h*128 + col)
#pragma unroll
  for (int r = 0; r < 8; ++r) {
    const int row = wm * 16 + hi * 8 + r;
    const float inv = 1.0f / lrow[row];
    float* dst = yws + ((size_t)b * 64 + row) * 2048 + h * 128 + wn * 64;
    dst[n]      = o0[r] * inv;
    dst[16 + n] = o1[r] * inv;
    dst[32 + n] = o2[r] * inv;
    dst[48 + n] = o3[r] * inv;
  }
}

extern "C" void kernel_launch(void* const* d_in, const int* in_sizes, int n_in,
                              void* d_out, int out_size, void* d_ws, size_t ws_size,
                              hipStream_t stream) {
  const float* x      = (const float*)d_in[0];
  const float* past_k = (const float*)d_in[1];
  const float* past_v = (const float*)d_in[2];
  const float* W_attn = (const float*)d_in[3];
  const float* W_proj = (const float*)d_in[4];

  float* y_out = (float*)d_out;                          // [8,64,2048]
  float* k_out = y_out + (size_t)8 * 64 * 2048;          // [8,16,4160,128]
  float* v_out = k_out + (size_t)8 * 16 * 4160 * 128;    // [8,16,4160,128]

  float* q_ws = (float*)d_ws;               // 512*2048 f32, scaled q (head layout)
  float* y_ws = q_ws + (size_t)512 * 2048;  // 512*2048 f32, attention output

  // 1) KV-cache concat (rows 0..4095): pure HBM streaming
  copy_cache_kernel<<<65536, 256, 0, stream>>>((const float4*)past_k, (float4*)k_out);
  copy_cache_kernel<<<65536, 256, 0, stream>>>((const float4*)past_v, (float4*)v_out);

  // 2) fused QKV projection: q -> ws (scaled), k_new/v_new -> rows 4096..4159
  gemm_bf16_kernel<<<dim3(48, 8), 256, 0, stream>>>(x, W_attn, 2048, 6144, 0,
                                                    q_ws, k_out, v_out, nullptr);

  // 3) flash attention over full 4160-key cache (no mask, per reference)
  attn_kernel<<<128, 256, 0, stream>>>(q_ws, k_out, v_out, y_ws);

  // 4) output projection
  gemm_bf16_kernel<<<dim3(16, 8), 256, 0, stream>>>(y_ws, W_proj, 2048, 2048, 1,
                                                    nullptr, nullptr, nullptr, y_out);
}
